// MLAAttention_30657476559556
// MI455X (gfx1250) — compile-verified, matrix operand order checked by reading, not run on hardware
//
#include <hip/hip_runtime.h>
#include <cstdint>
#include <cstddef>

// Problem constants (match reference)
#define BB    2
#define SS    2048
#define HIDD  2048
#define HH    16
#define HKV_  4
#define DD    128
#define LL    64
// N_REP = HH/HKV_ = 4 ; scale = 1/sqrt(LL) = 0.125

typedef __bf16 bf16;
typedef __attribute__((ext_vector_type(16))) __bf16 v16bf;
typedef __attribute__((ext_vector_type(8)))  __bf16 v8bf;
typedef __attribute__((ext_vector_type(8)))  float   v8f;

// gfx1250 async global->LDS copy path (ASYNCcnt-tracked), guarded so the file
// still compiles if this toolchain doesn't declare the builtins.
#if __has_builtin(__builtin_amdgcn_global_load_async_to_lds_b128) && \
    __has_builtin(__builtin_amdgcn_s_wait_asynccnt)
#define HAS_ASYNC_LDS 1
#else
#define HAS_ASYNC_LDS 0
#endif

#if HAS_ASYNC_LDS
// Builtin signature (from hipcc diagnostic): first param is
// 'int __attribute__((vector_size(16))) addrspace(1)*', second is the LDS
// (addrspace(3)) counterpart, then immediate offset and cpol.
typedef int v4i_vs __attribute__((vector_size(16)));
typedef __attribute__((address_space(1))) v4i_vs* g_v4i_p;
typedef __attribute__((address_space(3))) v4i_vs* l_v4i_p;
__device__ __forceinline__ void async_cp16(const void* gsrc, void* ldst) {
  __builtin_amdgcn_global_load_async_to_lds_b128(
      (g_v4i_p)gsrc, (l_v4i_p)ldst, 0, 0);
}
#endif

// Build a 16x32 bf16 A/B fragment for v_wmma_f32_16x16x32_bf16 from LDS/global.
// Per CDNA5 ISA: lane<16 holds K={0..7,16..23}, lane>=16 holds K={8..15,24..31},
// i.e. two contiguous 8-element (16B) chunks at +kb and +kb+16.
__device__ __forceinline__ v16bf ld_frag(const bf16* p) {
  v8bf lo = *(const v8bf*)(p);
  v8bf hi = *(const v8bf*)(p + 16);
  return __builtin_shufflevector(lo, hi, 0,1,2,3,4,5,6,7,8,9,10,11,12,13,14,15);
}

// ---------------------------------------------------------------------------
// Generic bf16 GEMM: C[M,N] (+)= A[M,K] * B[K,N], row-major, f32 accumulate.
// BLOCK_M=128, BLOCK_N=64, BLOCK_K=64. 256 threads = 8 waves, each wave 32x32.
// B is transposed into LDS so fragments are contiguous-K loads.
// Requires M%128==0, N%64==0, K%64==0 (true for all call sites here).
// ---------------------------------------------------------------------------
__global__ void __launch_bounds__(256)
gemm_bf16_wmma(const bf16* __restrict__ A, const bf16* __restrict__ B,
               void* __restrict__ Cv, int M, int N, int K, int outbf)
{
  __shared__ __align__(16) bf16 sA[128][72];
  __shared__ __align__(16) bf16 sBT[64][72];

  const int tid   = threadIdx.x;
  const int lane  = tid & 31;
  const int wave  = tid >> 5;
  const int laneN = lane & 15;
  const int hl    = lane >> 4;           // which K-half this lane holds
  const int wm    = wave & 3;            // 4 waves along M
  const int wn    = wave >> 2;           // 2 waves along N
  const int bm    = blockIdx.y * 128;
  const int bn    = blockIdx.x * 64;

  v8f acc[2][2] = {};

  const int nk = K >> 6;
  for (int kt = 0; kt < nk; ++kt) {
    // --- stage A tile: 128x64 bf16, 1024 16B-chunks over 256 threads ---
#if HAS_ASYNC_LDS
#pragma unroll
    for (int i = 0; i < 4; ++i) {
      int c   = tid + i * 256;
      int r   = c >> 3;
      int col = (c & 7) << 3;
      async_cp16(A + (size_t)(bm + r) * K + (size_t)kt * 64 + col, &sA[r][col]);
    }
#else
    {
      v8bf va[4];
#pragma unroll
      for (int i = 0; i < 4; ++i) {
        int c   = tid + i * 256;
        int r   = c >> 3;
        int col = (c & 7) << 3;
        va[i] = *(const v8bf*)(A + (size_t)(bm + r) * K + (size_t)kt * 64 + col);
      }
#pragma unroll
      for (int i = 0; i < 4; ++i) {
        int c   = tid + i * 256;
        int r   = c >> 3;
        int col = (c & 7) << 3;
        *(v8bf*)&sA[r][col] = va[i];
      }
    }
#endif
    // --- stage B tile transposed: sBT[n][k] = B[k][n] (register-staged) ---
    {
      v8bf vb[2];
#pragma unroll
      for (int i = 0; i < 2; ++i) {
        int c   = tid + i * 256;
        int kr  = c >> 3;
        int col = (c & 7) << 3;
        vb[i] = *(const v8bf*)(B + (size_t)(kt * 64 + kr) * N + bn + col);
      }
#pragma unroll
      for (int i = 0; i < 2; ++i) {
        int c   = tid + i * 256;
        int kr  = c >> 3;
        int col = (c & 7) << 3;
#pragma unroll
        for (int j = 0; j < 8; ++j) sBT[col + j][kr] = vb[i][j];
      }
    }
    if (kt + 1 < nk) {
      __builtin_prefetch(A + (size_t)(bm + (tid >> 1)) * K + (size_t)(kt + 1) * 64, 0, 0);
    }
#if HAS_ASYNC_LDS
    __builtin_amdgcn_s_wait_asynccnt(0);
#endif
    __syncthreads();

    // --- two WMMA k-steps of 32 ---
#pragma unroll
    for (int ks = 0; ks < 2; ++ks) {
      const int kb = ks * 32 + (hl ? 8 : 0);
      v16bf af[2], bfr[2];
#pragma unroll
      for (int mi = 0; mi < 2; ++mi)
        af[mi] = ld_frag(&sA[wm * 32 + mi * 16 + laneN][kb]);
#pragma unroll
      for (int ni = 0; ni < 2; ++ni)
        bfr[ni] = ld_frag(&sBT[wn * 32 + ni * 16 + laneN][kb]);
#pragma unroll
      for (int mi = 0; mi < 2; ++mi)
#pragma unroll
        for (int ni = 0; ni < 2; ++ni)
          acc[mi][ni] = __builtin_amdgcn_wmma_f32_16x16x32_bf16(
              false, af[mi], false, bfr[ni], (short)0, acc[mi][ni], false, false);
    }
    __syncthreads();
  }

  // --- epilogue: C layout = VGPR r -> row r (+8 for lane-half 1), lane%16 -> col ---
#pragma unroll
  for (int mi = 0; mi < 2; ++mi)
#pragma unroll
    for (int ni = 0; ni < 2; ++ni)
#pragma unroll
      for (int r = 0; r < 8; ++r) {
        int row = bm + wm * 32 + mi * 16 + r + hl * 8;
        int col = bn + wn * 32 + ni * 16 + laneN;
        float v = acc[mi][ni][r];
        if (outbf) ((bf16*)Cv)[(size_t)row * N + col] = (bf16)v;
        else       ((float*)Cv)[(size_t)row * N + col] = v;
      }
}

// ---------------------------------------------------------------------------
// Flash attention over latent dim L=64, GQA (4 q-heads per kv-head), causal.
// Grid: (S/64, H, B). Block: 128 threads = 4 waves, wave w owns 16 q rows.
// q_l/k_l/v_l are bf16 in (b, s, h, l) / (b, s, hkv, l) layout.
// Output bf16 in (b, s, h, l) layout (== A matrix of the final projection).
// ---------------------------------------------------------------------------
__global__ void __launch_bounds__(128)
mla_attn_wmma(const bf16* __restrict__ qlp, const bf16* __restrict__ klp,
              const bf16* __restrict__ vlp, bf16* __restrict__ outp)
{
  __shared__ __align__(16) bf16 sK[64][72];    // [key][l]  -> B-frag for QK^T
  __shared__ __align__(16) bf16 sVT[64][72];   // [l][key]  -> B-frag for PV
  __shared__ __align__(16) bf16 sP[4][16][72]; // per-wave P staging (A-frag relayout)

  const int tid   = threadIdx.x;
  const int lane  = tid & 31;
  const int wave  = tid >> 5;
  const int laneN = lane & 15;
  const int hl    = lane >> 4;
  const int bx    = blockIdx.x;           // q tile index
  const int h     = blockIdx.y;
  const int b     = blockIdx.z;
  const int hk    = h >> 2;               // kv head (N_REP = 4)
  const int q0    = bx * 64 + wave * 16;  // first q row of this wave

  // Q fragments for this wave's 16 rows, kept in registers all kernel.
  v16bf qf[2];
  {
    const bf16* qr = qlp + (((size_t)(b * SS + q0 + laneN)) * HH + h) * LL;
#pragma unroll
    for (int ks = 0; ks < 2; ++ks)
      qf[ks] = ld_frag(qr + ks * 32 + (hl ? 8 : 0));
  }

  v8f oacc[4] = {};
  float m_run[8], l_run[8];
#pragma unroll
  for (int r = 0; r < 8; ++r) { m_run[r] = -1e30f; l_run[r] = 0.0f; }

  const int nkt = bx + 1;                 // causal: only tiles up to diagonal
  for (int kt = 0; kt < nkt; ++kt) {
    // --- cooperative load of K (straight / async) and V (transposed) tiles ---
#if HAS_ASYNC_LDS
#pragma unroll
    for (int i = 0; i < 4; ++i) {
      int c   = tid + i * 128;            // 0..511, 16B chunks of 64x64
      int key = c >> 3;
      int col = (c & 7) << 3;
      size_t base = (((size_t)(b * SS + kt * 64 + key)) * HKV_ + hk) * LL + col;
      async_cp16(klp + base, &sK[key][col]);
    }
#else
    {
      v8bf kv[4];
#pragma unroll
      for (int i = 0; i < 4; ++i) {
        int c   = tid + i * 128;
        int key = c >> 3;
        int col = (c & 7) << 3;
        size_t base = (((size_t)(b * SS + kt * 64 + key)) * HKV_ + hk) * LL + col;
        kv[i] = *(const v8bf*)(klp + base);
      }
#pragma unroll
      for (int i = 0; i < 4; ++i) {
        int c   = tid + i * 128;
        int key = c >> 3;
        int col = (c & 7) << 3;
        *(v8bf*)&sK[key][col] = kv[i];
      }
    }
#endif
    {
      v8bf vv[4];
#pragma unroll
      for (int i = 0; i < 4; ++i) {
        int c   = tid + i * 128;
        int key = c >> 3;
        int col = (c & 7) << 3;
        size_t base = (((size_t)(b * SS + kt * 64 + key)) * HKV_ + hk) * LL + col;
        vv[i] = *(const v8bf*)(vlp + base);
      }
#pragma unroll
      for (int i = 0; i < 4; ++i) {
        int c   = tid + i * 128;
        int key = c >> 3;
        int col = (c & 7) << 3;
#pragma unroll
        for (int j = 0; j < 8; ++j) sVT[col + j][key] = vv[i][j];
      }
    }
#if HAS_ASYNC_LDS
    __builtin_amdgcn_s_wait_asynccnt(0);
#endif
    __syncthreads();

    // --- scores: S = Q (16xL) * K^T (Lx64), 4 n-tiles of 16 keys ---
    v8f acc[4] = {};
#pragma unroll
    for (int ks = 0; ks < 2; ++ks) {
      const int kb = ks * 32 + (hl ? 8 : 0);
#pragma unroll
      for (int ni = 0; ni < 4; ++ni) {
        v16bf bfr = ld_frag(&sK[ni * 16 + laneN][kb]);
        acc[ni] = __builtin_amdgcn_wmma_f32_16x16x32_bf16(
            false, qf[ks], false, bfr, (short)0, acc[ni], false, false);
      }
    }

    // --- online softmax: rows live per accumulator-VGPR r, per lane-half ---
#pragma unroll
    for (int r = 0; r < 8; ++r) {
      const int row = q0 + r + hl * 8;
      float mx = -1e30f;
#pragma unroll
      for (int ni = 0; ni < 4; ++ni) {
        int key = kt * 64 + ni * 16 + laneN;
        float sv = acc[ni][r] * 0.125f + ((key <= row) ? 0.0f : -1e30f);
        acc[ni][r] = sv;
        mx = fmaxf(mx, sv);
      }
#pragma unroll
      for (int mk = 1; mk < 16; mk <<= 1)     // stays within 16-lane half
        mx = fmaxf(mx, __shfl_xor(mx, mk, 32));
      const float mnew  = fmaxf(m_run[r], mx);
      const float alpha = __expf(m_run[r] - mnew);
      float rs = 0.0f;
#pragma unroll
      for (int ni = 0; ni < 4; ++ni) {
        float p = __expf(acc[ni][r] - mnew);
        acc[ni][r] = p;
        rs += p;
      }
#pragma unroll
      for (int mk = 1; mk < 16; mk <<= 1)
        rs += __shfl_xor(rs, mk, 32);
      l_run[r] = l_run[r] * alpha + rs;
      m_run[r] = mnew;
#pragma unroll
      for (int li = 0; li < 4; ++li) oacc[li][r] *= alpha;
      // spill P (bf16) to per-wave LDS for A-fragment relayout
#pragma unroll
      for (int ni = 0; ni < 4; ++ni)
        sP[wave][r + hl * 8][ni * 16 + laneN] = (bf16)acc[ni][r];
    }

    // --- O += P (16x64) * V (64xL) ---
#pragma unroll
    for (int ks = 0; ks < 2; ++ks) {
      const int kb = ks * 32 + (hl ? 8 : 0);
      v16bf pf = ld_frag(&sP[wave][laneN][kb]);
#pragma unroll
      for (int li = 0; li < 4; ++li) {
        v16bf vf = ld_frag(&sVT[li * 16 + laneN][kb]);
        oacc[li] = __builtin_amdgcn_wmma_f32_16x16x32_bf16(
            false, pf, false, vf, (short)0, oacc[li], false, false);
      }
    }
    __syncthreads();
  }

  // --- normalize and write (b, s, h, l) bf16 ---
#pragma unroll
  for (int r = 0; r < 8; ++r) {
    const int row = q0 + r + hl * 8;
    const float inv = 1.0f / l_run[r];
#pragma unroll
    for (int li = 0; li < 4; ++li)
      outp[(((size_t)(b * SS + row)) * HH + h) * LL + li * 16 + laneN] =
          (bf16)(oacc[li][r] * inv);
  }
}

// ---------------------------------------------------------------------------
// Elementwise helpers
// ---------------------------------------------------------------------------
__global__ void cvt_f32_bf16(const float* __restrict__ in, bf16* __restrict__ out, int n)
{
  int i = blockIdx.x * blockDim.x + threadIdx.x;
  if (i < n) out[i] = (bf16)in[i];
}

// Gather only the first L=64 rows of each head's D=128 block of Wo (zero-pad
// in the reference makes the other 64 rows dead) -> Wo_eff (H*L x HID) bf16.
__global__ void gather_wo_bf16(const float* __restrict__ Wo, bf16* __restrict__ out)
{
  const int n = HH * LL * HIDD;
  int i = blockIdx.x * blockDim.x + threadIdx.x;
  if (i >= n) return;
  int col = i % HIDD;
  int row = i / HIDD;          // h*64 + l
  int h = row >> 6, l = row & 63;
  out[i] = (bf16)Wo[((size_t)(h * DD + l)) * HIDD + col];
}

// RoPE (half-rotation, theta=10000, D=128) fused with f32->bf16 convert.
// in/out layout: (row, head, d) with row = b*S + s, position = s.
__global__ void rope_cvt_bf16(const float* __restrict__ in, bf16* __restrict__ out,
                              int rows, int heads, int doRope)
{
  int idx = blockIdx.x * blockDim.x + threadIdx.x;
  int total = rows * heads * 64;
  if (idx >= total) return;
  int i   = idx & 63;
  int hh  = (idx >> 6) % heads;
  int row = idx / (heads * 64);
  const float* p = in  + ((size_t)row * heads + hh) * DD;
  bf16*        o = out + ((size_t)row * heads + hh) * DD;
  float t1 = p[i], t2 = p[i + 64];
  if (doRope) {
    int pos = row % SS;                               // position_ids = arange(S)
    // inv_freq = theta^(-2i/128) ; ln(10000) = 9.2103403719761836
    float freq = __expf(-9.2103403719761836f * (float)(2 * i) * (1.0f / 128.0f));
    float ang = (float)pos * freq;
    float s, c;
    __sincosf(ang, &s, &c);
    o[i]      = (bf16)(t1 * c - t2 * s);
    o[i + 64] = (bf16)(t2 * c + t1 * s);
  } else {
    o[i]      = (bf16)t1;
    o[i + 64] = (bf16)t2;
  }
}

// ---------------------------------------------------------------------------
// Host-side orchestration
// ---------------------------------------------------------------------------
extern "C" void kernel_launch(void* const* d_in, const int* in_sizes, int n_in,
                              void* d_out, int out_size, void* d_ws, size_t ws_size,
                              hipStream_t stream)
{
  (void)in_sizes; (void)n_in; (void)out_size; (void)ws_size;
  const float* x   = (const float*)d_in[0];
  // d_in[1] position_ids (arange) and d_in[2] attention_mask (causal) are
  // reproduced analytically in-kernel.
  const float* Wq  = (const float*)d_in[3];
  const float* Wk  = (const float*)d_in[4];
  const float* Wv  = (const float*)d_in[5];
  const float* Wql = (const float*)d_in[6];
  const float* Wkl = (const float*)d_in[7];
  const float* Wvl = (const float*)d_in[8];
  const float* Wo  = (const float*)d_in[9];

  const size_t M = (size_t)BB * SS;             // 4096 token rows
  char* ws = (char*)d_ws;
  size_t cur = 0;
  auto alloc = [&](size_t bytes) {
    size_t o = cur;
    cur += (bytes + 255) & ~(size_t)255;
    return o;
  };

  const size_t o_xbf = alloc(M * HIDD * 2);                 // x bf16
  const size_t o_wq  = alloc((size_t)HIDD * HIDD * 2);      // Wq bf16
  const size_t o_wk  = alloc((size_t)HIDD * HKV_ * DD * 2); // Wk bf16
  const size_t o_wv  = alloc((size_t)HIDD * HKV_ * DD * 2); // Wv bf16
  const size_t o_wql = alloc((size_t)DD * LL * 2);
  const size_t o_wkl = alloc((size_t)DD * LL * 2);
  const size_t o_wvl = alloc((size_t)DD * LL * 2);
  const size_t o_wo  = alloc((size_t)HH * LL * HIDD * 2);   // Wo_eff bf16
  const size_t o_q32 = alloc(M * HIDD * 4);                 // q f32 (pre-RoPE)
  const size_t o_k32 = alloc(M * HKV_ * DD * 4);
  const size_t o_v32 = alloc(M * HKV_ * DD * 4);
  const size_t o_ql  = alloc(M * HH * LL * 2);              // q_l bf16 (b,s,h,l)
  const size_t o_kl  = alloc(M * HKV_ * LL * 2);
  const size_t o_vl  = alloc(M * HKV_ * LL * 2);
  // Buffer reuse (lifetimes are disjoint):
  const size_t o_qbf  = o_xbf;                        // q bf16 over x bf16
  const size_t o_kbf  = o_wq;                         // k bf16 over Wq bf16
  const size_t o_vbf  = o_wq + M * HKV_ * DD * 2;     // v bf16 in Wq region too
  const size_t o_attn = o_q32;                        // attn out bf16 over q f32

  bf16* xbf   = (bf16*)(ws + o_xbf);
  bf16* wqbf  = (bf16*)(ws + o_wq);
  bf16* wkbf  = (bf16*)(ws + o_wk);
  bf16* wvbf  = (bf16*)(ws + o_wv);
  bf16* wqlbf = (bf16*)(ws + o_wql);
  bf16* wklbf = (bf16*)(ws + o_wkl);
  bf16* wvlbf = (bf16*)(ws + o_wvl);
  bf16* wobf  = (bf16*)(ws + o_wo);
  float* q32  = (float*)(ws + o_q32);
  float* k32  = (float*)(ws + o_k32);
  float* v32  = (float*)(ws + o_v32);
  bf16* qbf   = (bf16*)(ws + o_qbf);
  bf16* kbf   = (bf16*)(ws + o_kbf);
  bf16* vbf   = (bf16*)(ws + o_vbf);
  bf16* qlp   = (bf16*)(ws + o_ql);
  bf16* klp   = (bf16*)(ws + o_kl);
  bf16* vlp   = (bf16*)(ws + o_vl);
  bf16* attnb = (bf16*)(ws + o_attn);

  const int T = 256;
  auto nb = [](size_t n) { return (unsigned)((n + 255) / 256); };

  // 1) precision staging
  cvt_f32_bf16<<<nb(M * HIDD), T, 0, stream>>>(x, xbf, (int)(M * HIDD));
  cvt_f32_bf16<<<nb((size_t)HIDD * HIDD), T, 0, stream>>>(Wq, wqbf, HIDD * HIDD);
  cvt_f32_bf16<<<nb((size_t)HIDD * HKV_ * DD), T, 0, stream>>>(Wk, wkbf, HIDD * HKV_ * DD);
  cvt_f32_bf16<<<nb((size_t)HIDD * HKV_ * DD), T, 0, stream>>>(Wv, wvbf, HIDD * HKV_ * DD);
  cvt_f32_bf16<<<nb((size_t)DD * LL), T, 0, stream>>>(Wql, wqlbf, DD * LL);
  cvt_f32_bf16<<<nb((size_t)DD * LL), T, 0, stream>>>(Wkl, wklbf, DD * LL);
  cvt_f32_bf16<<<nb((size_t)DD * LL), T, 0, stream>>>(Wvl, wvlbf, DD * LL);
  gather_wo_bf16<<<nb((size_t)HH * LL * HIDD), T, 0, stream>>>(Wo, wobf);

  // 2) QKV projections (WMMA)
  gemm_bf16_wmma<<<dim3(HIDD / 64, (unsigned)(M / 128)), 256, 0, stream>>>(
      xbf, wqbf, q32, (int)M, HIDD, HIDD, 0);
  gemm_bf16_wmma<<<dim3((HKV_ * DD) / 64, (unsigned)(M / 128)), 256, 0, stream>>>(
      xbf, wkbf, k32, (int)M, HKV_ * DD, HIDD, 0);
  gemm_bf16_wmma<<<dim3((HKV_ * DD) / 64, (unsigned)(M / 128)), 256, 0, stream>>>(
      xbf, wvbf, v32, (int)M, HKV_ * DD, HIDD, 0);

  // 3) RoPE + bf16 (q,k); plain convert (v)
  rope_cvt_bf16<<<nb(M * HH * 64), T, 0, stream>>>(q32, qbf, (int)M, HH, 1);
  rope_cvt_bf16<<<nb(M * HKV_ * 64), T, 0, stream>>>(k32, kbf, (int)M, HKV_, 1);
  rope_cvt_bf16<<<nb(M * HKV_ * 64), T, 0, stream>>>(v32, vbf, (int)M, HKV_, 0);

  // 4) latent projections (WMMA, bf16 out), rows = (b,s,h) so output is (b,s,h,l)
  gemm_bf16_wmma<<<dim3(1, (unsigned)(M * HH / 128)), 256, 0, stream>>>(
      qbf, wqlbf, qlp, (int)(M * HH), LL, DD, 1);
  gemm_bf16_wmma<<<dim3(1, (unsigned)(M * HKV_ / 128)), 256, 0, stream>>>(
      kbf, wklbf, klp, (int)(M * HKV_), LL, DD, 1);
  gemm_bf16_wmma<<<dim3(1, (unsigned)(M * HKV_ / 128)), 256, 0, stream>>>(
      vbf, wvlbf, vlp, (int)(M * HKV_), LL, DD, 1);

  // 5) causal GQA flash attention at L=64 (WMMA QK^T and PV)
  mla_attn_wmma<<<dim3(SS / 64, HH, BB), 128, 0, stream>>>(qlp, klp, vlp, attnb);

  // 6) output projection through gathered Wo (K = H*L = 1024) -> f32 d_out
  gemm_bf16_wmma<<<dim3(HIDD / 64, (unsigned)(M / 128)), 256, 0, stream>>>(
      attnb, wobf, (float*)d_out, (int)M, HIDD, HH * LL, 0);
}
